// NarFormer2_8787503088295
// MI455X (gfx1250) — matrix operand order repaired and checked by reading, not compiled, not run on hardware
//
#include <hip/hip_runtime.h>
#include <hip/hip_bf16.h>

#define B_ 32
#define N_ 1024
#define F0_ 128
#define H_ 256
#define G_ 8
#define FH_ 1024
#define ROWS (B_*N_)   /* 32768 */

// LDS staging geometry for WMMA B-panels: 128 rows x 64 halves per chunk,
// rows padded to 72 halves (144B) to spread DS bank accesses.
#define CHK 64
#define LDSROW 72

typedef __attribute__((ext_vector_type(16))) _Float16 v16h;
typedef __attribute__((ext_vector_type(8)))  _Float16 v8h;
typedef __attribute__((ext_vector_type(8)))  float    v8f;

__device__ __forceinline__ float sigmoidf_(float x){ return 1.f/(1.f+__expf(-x)); }

// CDNA5 async global->LDS copy (ASYNCcnt-tracked). vdst = per-lane LDS byte
// address (low 32 bits of the generic __shared__ pointer), vaddr = global addr.
__device__ __forceinline__ void async_ld_b128(const void* g, void* lds){
  unsigned lv = (unsigned)(unsigned long long)lds;
  asm volatile("global_load_async_to_lds_b128 %0, %1, off"
               :: "v"(lv), "v"(g) : "memory");
}
#define ASYNC_WAIT(n) asm volatile("s_wait_asynccnt " #n ::: "memory")

// Cooperatively stage a 128 x CHK(halves) B-panel chunk into LDS (row-major,
// LDSROW pitch). 256 threads, 4 async b128 loads each (64B per thread).
__device__ __forceinline__ void stage_chunk(const _Float16* gsrc, int ldg, _Float16* sbuf){
  int t = threadIdx.x;
  int row = t >> 1, part = t & 1;
  const _Float16* g = gsrc + (size_t)row*ldg + part*32;
  _Float16* l = sbuf + row*LDSROW + part*32;
#pragma unroll
  for (int j=0;j<4;j++) async_ld_b128(g + j*8, l + j*8);
}

// A fragment 16x32 f16 (row-major source, leading dim ld, p -> (m0,k0)).
__device__ __forceinline__ v16h load_a_frag(const _Float16* p, int ld){
  int lane = threadIdx.x & 31;
  int m = lane & 15, hh = lane >> 4;
  const _Float16* r = p + (size_t)m*ld + 16*hh;
  v8h lo = *(const v8h*)(r);
  v8h hi = *(const v8h*)(r + 8);
  v16h a;
#pragma unroll
  for (int i=0;i<8;i++){ a[i]=lo[i]; a[i+8]=hi[i]; }
  return a;
}

// B fragment 32x16 f16 from N-major / K-contiguous ("nk") matrix in global mem
__device__ __forceinline__ v16h load_b_frag(const _Float16* p, int ld){
  int lane = threadIdx.x & 31;
  int n = lane & 15, hh = lane >> 4;
  return *(const v16h*)(p + (size_t)n*ld + 16*hh);
}

// B fragment from an LDS-staged panel (16B-aligned ds_load_b128 pairs)
__device__ __forceinline__ v16h load_b_frag_lds(const _Float16* p, int ld){
  int lane = threadIdx.x & 31;
  int n = lane & 15, hh = lane >> 4;
  const _Float16* r = p + n*ld + 16*hh;
  v8h lo = *(const v8h*)(r);
  v8h hi = *(const v8h*)(r + 8);
  v16h b;
#pragma unroll
  for (int i=0;i<8;i++){ b[i]=lo[i]; b[i+8]=hi[i]; }
  return b;
}

__device__ __forceinline__ v8f wmma32(v16h a, v16h b, v8f c){
  return __builtin_amdgcn_wmma_f32_16x16x32_f16(false, a, false, b, (short)0, c, false, false);
}

// ---------------- elementwise / setup kernels ----------------

__global__ void k_deg(const float* __restrict__ A, float* __restrict__ deg){
  int wave = threadIdx.x >> 5, lane = threadIdx.x & 31;
  int row = blockIdx.x*8 + wave;
  const float* p = A + (size_t)row * N_;
  float s = 0.f;
  for (int i = lane; i < N_; i += 32) s += p[i];
#pragma unroll
  for (int o=16;o;o>>=1) s += __shfl_xor(s, o);
  if (lane==0) deg[row] = s;
}

__global__ void k_zero(float* p, int n){
  int i = blockIdx.x*blockDim.x + threadIdx.x;
  if (i<n) p[i]=0.f;
}

__global__ void k_f32_to_f16(const float* __restrict__ s, _Float16* __restrict__ d, int n){
  int i = blockIdx.x*blockDim.x + threadIdx.x;
  if (i<n) d[i] = (_Float16)s[i];
}

// (G,I,O) -> (G,O,I) f16
__global__ void k_glt_wT(const float* __restrict__ w, _Float16* __restrict__ wt, int I, int O){
  int idx = blockIdx.x*blockDim.x + threadIdx.x;
  int tot = G_*I*O; if (idx>=tot) return;
  int g = idx/(I*O); int rem = idx - g*I*O; int o = rem / I; int i = rem - o*I;
  wt[idx] = (_Float16)w[((size_t)g*I + i)*O + o];
}

__global__ void k_gate(const float* __restrict__ xin, const float* __restrict__ deg,
                       const float* __restrict__ dw, const float* __restrict__ db,
                       _Float16* __restrict__ xg, _Float16* __restrict__ xgT, int d){
  int idx = blockIdx.x*blockDim.x + threadIdx.x;
  int row = idx / d, i = idx - row*d;
  int b = row >> 10, n = row & (N_-1);
  float v = xin[idx] * sigmoidf_(deg[row]*dw[i] + db[i]);
  xg[idx] = (_Float16)v;
  xgT[((size_t)b*d + i)*N_ + n] = (_Float16)v;
}

// ---------------- WMMA GEMM kernels ----------------

// qk = sigmoid(xg @ qw^T + qb)  M=32768,K=d,N=d ; B panel (128 x d) LDS-staged
__global__ void k_qk(const _Float16* __restrict__ xg, const _Float16* __restrict__ qwT,
                     const float* __restrict__ qb, _Float16* __restrict__ qk, int d){
  __shared__ _Float16 sb[2][128*LDSROW];
  int wave = threadIdx.x >> 5;
  int m0 = blockIdx.x * 16;
  int np = blockIdx.y * 128;          // panel base
  int n0 = np + wave * 16;
  v8f acc = {};
  const _Float16* ap = xg  + (size_t)m0 * d;
  const _Float16* bp = qwT + (size_t)np * d;
  int nch = d / CHK;
  stage_chunk(bp, d, sb[0]);
  for (int c=0;c<nch;c++){
    if (c+1<nch){ stage_chunk(bp + (c+1)*CHK, d, sb[(c+1)&1]); ASYNC_WAIT(4); }
    else        { ASYNC_WAIT(0); }
    __syncthreads();
    const _Float16* sc = sb[c&1] + wave*16*LDSROW;
#pragma unroll
    for (int ks=0;ks<2;ks++){
      v16h a = load_a_frag(ap + c*CHK + ks*32, d);
      acc = wmma32(a, load_b_frag_lds(sc + ks*32, LDSROW), acc);
    }
    __syncthreads();
  }
  int lane=threadIdx.x&31, cn=lane&15, ch=lane>>4;
#pragma unroll
  for (int r=0;r<8;r++){
    int m = m0 + r + 8*ch, n = n0 + cn;
    qk[(size_t)m*d + n] = (_Float16)sigmoidf_(acc[r] + qb[n]);
  }
}

// S = (qk qk^T)*scale*A (f16), rowsum += sum_j S  (128x128 tile; B panel staged)
__global__ void k_score(const _Float16* __restrict__ qk, const float* __restrict__ A,
                        _Float16* __restrict__ S, float* __restrict__ rowsum,
                        float scale, int d){
  __shared__ _Float16 sb[2][128*LDSROW];
  int b = blockIdx.x, mi = blockIdx.y, ni = blockIdx.z;
  int wave = threadIdx.x>>5;
  int m0 = mi*128 + wave*16;
  int n0 = ni*128;
  const _Float16* qB = qk + (size_t)b*N_*d;
  v8f acc[8] = {};
  const _Float16* ap = qB + (size_t)m0*d;
  const _Float16* bp = qB + (size_t)n0*d;
  int nch = d / CHK;
  stage_chunk(bp, d, sb[0]);
  for (int c=0;c<nch;c++){
    if (c+1<nch){ stage_chunk(bp + (c+1)*CHK, d, sb[(c+1)&1]); ASYNC_WAIT(4); }
    else        { ASYNC_WAIT(0); }
    __syncthreads();
    const _Float16* sc = sb[c&1];
#pragma unroll
    for (int ks=0;ks<2;ks++){
      v16h a = load_a_frag(ap + c*CHK + ks*32, d);
#pragma unroll
      for (int t=0;t<8;t++)
        acc[t] = wmma32(a, load_b_frag_lds(sc + t*16*LDSROW + ks*32, LDSROW), acc[t]);
    }
    __syncthreads();
  }
  int lane=threadIdx.x&31, cn=lane&15, ch=lane>>4;
  float rs[8] = {0.f,0.f,0.f,0.f,0.f,0.f,0.f,0.f};
#pragma unroll
  for (int r=0;r<8;r++){
    int m = m0 + r + 8*ch;
    const float* Arow = A + ((size_t)b*N_ + m)*N_;
    _Float16*    Srow = S + ((size_t)b*N_ + m)*N_;
#pragma unroll
    for (int t=0;t<8;t++){
      int n = n0 + t*16 + cn;
      float s = acc[t][r] * scale * Arow[n];
      Srow[n] = (_Float16)s;
      rs[r] += s;
    }
  }
#pragma unroll
  for (int r=0;r<8;r++){
    float v = rs[r];
    v += __shfl_xor(v,1); v += __shfl_xor(v,2); v += __shfl_xor(v,4); v += __shfl_xor(v,8);
    if (cn==0) atomicAdd(rowsum + b*N_ + m0 + r + 8*ch, v);
  }
}

// Z1 = (S @ Xg) / (rowsum+eps)  K=1024; B panel (128 feats x 1024) LDS-staged
__global__ void k_attnx(const _Float16* __restrict__ S, const _Float16* __restrict__ xgT,
                        const float* __restrict__ rowsum, _Float16* __restrict__ Z1, int d){
  __shared__ _Float16 sb[2][128*LDSROW];
  int b = blockIdx.x, mi = blockIdx.y, fi = blockIdx.z;
  int wave = threadIdx.x>>5;
  int m0 = mi*128 + wave*16;
  int f0 = fi*128;
  const _Float16* Sp = S + ((size_t)b*N_ + m0)*N_;
  const _Float16* Bp = xgT + (size_t)b*d*N_ + (size_t)f0*N_;
  v8f acc[8] = {};
  stage_chunk(Bp, N_, sb[0]);
  for (int c=0;c<N_/CHK;c++){
    if (c+1<N_/CHK){ stage_chunk(Bp + (c+1)*CHK, N_, sb[(c+1)&1]); ASYNC_WAIT(4); }
    else           { ASYNC_WAIT(0); }
    __syncthreads();
    const _Float16* sc = sb[c&1];
#pragma unroll
    for (int ks=0;ks<2;ks++){
      v16h a = load_a_frag(Sp + c*CHK + ks*32, N_);
#pragma unroll
      for (int t=0;t<8;t++)
        acc[t] = wmma32(a, load_b_frag_lds(sc + t*16*LDSROW + ks*32, LDSROW), acc[t]);
    }
    __syncthreads();
  }
  int lane=threadIdx.x&31, cn=lane&15, ch=lane>>4;
#pragma unroll
  for (int r=0;r<8;r++){
    int m = m0 + r + 8*ch;
    float inv = 1.f/(rowsum[b*N_ + m] + 1e-6f);
    _Float16* Zr = Z1 + ((size_t)b*N_ + m)*d + f0;
#pragma unroll
    for (int t=0;t<8;t++) Zr[t*16 + cn] = (_Float16)(acc[t][r]*inv);
  }
}

// out = Z1@lw^T + xg@rw^T + lb; row-norm; relu -> h(f32); LayerNorm -> zln(f16)
__global__ void k_mix(const _Float16* __restrict__ Z1, const _Float16* __restrict__ xg,
                      const _Float16* __restrict__ lw, const _Float16* __restrict__ rw,
                      const float* __restrict__ lb, const float* __restrict__ lng,
                      const float* __restrict__ lnb, float* __restrict__ h,
                      _Float16* __restrict__ zln, int d){
  __shared__ float yt[16*H_];
  __shared__ float rnorm[16], rmean[16], rstdv[16];
  int wave = threadIdx.x>>5;
  int m0 = blockIdx.x*16;
  int n0 = wave*32;
  v8f a0 = {}, a1 = {};
  for (int k=0;k<d;k+=32){
    v16h a = load_a_frag(Z1 + (size_t)m0*d + k, d);
    a0 = wmma32(a, load_b_frag(lw + (size_t)n0*d + k, d), a0);
    a1 = wmma32(a, load_b_frag(lw + (size_t)(n0+16)*d + k, d), a1);
  }
  for (int k=0;k<d;k+=32){
    v16h a = load_a_frag(xg + (size_t)m0*d + k, d);
    a0 = wmma32(a, load_b_frag(rw + (size_t)n0*d + k, d), a0);
    a1 = wmma32(a, load_b_frag(rw + (size_t)(n0+16)*d + k, d), a1);
  }
  int lane=threadIdx.x&31, cn=lane&15, ch=lane>>4;
#pragma unroll
  for (int r=0;r<8;r++){
    int row = r + 8*ch;
    yt[row*H_ + n0 + cn]      = a0[r] + lb[n0+cn];
    yt[row*H_ + n0 + 16 + cn] = a1[r] + lb[n0+16+cn];
  }
  __syncthreads();
  int row = threadIdx.x >> 4, t16 = threadIdx.x & 15;
  float ss = 0.f;
  for (int j=t16;j<H_;j+=16){ float v = yt[row*H_+j]; ss += v*v; }
  ss += __shfl_xor(ss,1); ss += __shfl_xor(ss,2); ss += __shfl_xor(ss,4); ss += __shfl_xor(ss,8);
  if (t16==0) rnorm[row] = sqrtf(ss);
  __syncthreads();
  float inv = 1.f / fmaxf(rnorm[row], 1e-12f);
  float s1=0.f, s2=0.f;
  for (int j=t16;j<H_;j+=16){
    float v = fmaxf(yt[row*H_+j]*inv, 0.f);
    yt[row*H_+j] = v;
    h[(size_t)(m0+row)*H_ + j] = v;
    s1 += v; s2 += v*v;
  }
  s1 += __shfl_xor(s1,1); s1 += __shfl_xor(s1,2); s1 += __shfl_xor(s1,4); s1 += __shfl_xor(s1,8);
  s2 += __shfl_xor(s2,1); s2 += __shfl_xor(s2,2); s2 += __shfl_xor(s2,4); s2 += __shfl_xor(s2,8);
  if (t16==0){
    float mean = s1*(1.f/H_);
    rmean[row]=mean;
    rstdv[row]=rsqrtf(s2*(1.f/H_)-mean*mean + 1e-5f);
  }
  __syncthreads();
  float mean = rmean[row], rst = rstdv[row];
  for (int j=t16;j<H_;j+=16){
    float v = (yt[row*H_+j]-mean)*rst*lng[j] + lnb[j];
    zln[(size_t)(m0+row)*H_ + j] = (_Float16)v;
  }
}

// GLT up: per group g, [16x32]@[32x128] (K=32, one WMMA per tile),
// channel-shuffle + LayerNorm(128) + relu -> zup f16
__global__ void k_glt_up(const _Float16* __restrict__ z, const _Float16* __restrict__ wT,
                         const float* __restrict__ bias, const float* __restrict__ gam,
                         const float* __restrict__ bet, _Float16* __restrict__ zup){
  __shared__ float yt[16*FH_]; // 64KB
  int g = threadIdx.x>>5;      // wave == group
  int m0 = blockIdx.x*16;
  v8f acc[8] = {};
  v16h a = load_a_frag(z + (size_t)m0*H_ + g*32, H_);
#pragma unroll
  for (int t=0;t<8;t++)
    acc[t] = wmma32(a, load_b_frag(wT + ((size_t)g*128 + t*16)*32, 32), acc[t]);
  int lane=threadIdx.x&31, cn=lane&15, ch=lane>>4;
#pragma unroll
  for (int t=0;t<8;t++)
#pragma unroll
    for (int r=0;r<8;r++){
      int row=r+8*ch, o=t*16+cn;
      yt[row*FH_ + g*128 + o] = acc[t][r] + bias[g*128 + o];
    }
  __syncthreads();
  int inst = threadIdx.x >> 1, hf = threadIdx.x & 1;
  int row = inst >> 3, g2 = inst & 7;
  float s1=0.f, s2=0.f;
  for (int gg=hf*4; gg<hf*4+4; ++gg)
#pragma unroll
    for (int q=0;q<16;q++){
      float v = yt[row*FH_ + gg*128 + g2*16 + q]; s1+=v; s2+=v*v;
    }
  s1 += __shfl_xor(s1,1); s2 += __shfl_xor(s2,1);
  float mean = s1*(1.f/128.f);
  float rst = rsqrtf(s2*(1.f/128.f) - mean*mean + 1e-5f);
  for (int gg=hf*4; gg<hf*4+4; ++gg)
    for (int q=0;q<16;q++){
      int c2 = q*8 + gg;
      float v = (yt[row*FH_ + gg*128 + g2*16 + q] - mean)*rst*gam[c2] + bet[c2];
      zup[(size_t)(m0+row)*FH_ + g2*128 + c2] = (_Float16)fmaxf(v, 0.f);
    }
}

// GLT down + shuffle + LayerNorm(32) + residual: h += ls * z (in place)
__global__ void k_glt_dn(const _Float16* __restrict__ zup, const _Float16* __restrict__ wT,
                         const float* __restrict__ bias, const float* __restrict__ gam,
                         const float* __restrict__ bet, const float* __restrict__ ls,
                         float* __restrict__ h){
  __shared__ float yt[16*H_];
  int g = threadIdx.x>>5;
  int m0 = blockIdx.x*16;
  v8f a0={}, a1={};
  for (int k=0;k<128;k+=32){
    v16h a = load_a_frag(zup + (size_t)m0*FH_ + g*128 + k, FH_);
    a0 = wmma32(a, load_b_frag(wT + ((size_t)g*32 +  0)*128 + k, 128), a0);
    a1 = wmma32(a, load_b_frag(wT + ((size_t)g*32 + 16)*128 + k, 128), a1);
  }
  int lane=threadIdx.x&31, cn=lane&15, ch=lane>>4;
#pragma unroll
  for (int r=0;r<8;r++){
    int row=r+8*ch;
    yt[row*H_ + g*32 + cn]      = a0[r] + bias[g*32 + cn];
    yt[row*H_ + g*32 + 16 + cn] = a1[r] + bias[g*32 + 16 + cn];
  }
  __syncthreads();
  int inst = threadIdx.x>>1, hf = threadIdx.x&1;
  int row = inst>>3, g2 = inst&7;
  float s1=0.f,s2=0.f;
  for (int gg=hf*4; gg<hf*4+4; ++gg)
#pragma unroll
    for (int q=0;q<4;q++){
      float v = yt[row*H_ + gg*32 + g2*4 + q]; s1+=v; s2+=v*v;
    }
  s1 += __shfl_xor(s1,1); s2 += __shfl_xor(s2,1);
  float mean = s1*(1.f/32.f);
  float rst = rsqrtf(s2*(1.f/32.f) - mean*mean + 1e-5f);
  for (int gg=hf*4; gg<hf*4+4; ++gg)
    for (int q=0;q<4;q++){
      int c2 = q*8 + gg;
      float v = (yt[row*H_ + gg*32 + g2*4 + q] - mean)*rst*gam[c2] + bet[c2];
      int f = g2*32 + c2;
      size_t ix = (size_t)(m0+row)*H_ + f;
      h[ix] = h[ix] + ls[f]*v;
    }
}

// ---------------- pooling + head ----------------

__global__ void k_pool(const float* __restrict__ h, float* __restrict__ f){
  int b = blockIdx.x, j = threadIdx.x;
  float s=0.f;
  for (int n=0;n<N_;n++) s += h[((size_t)b*N_+n)*H_ + j];
  f[b*H_+j] = s;
}

__global__ void k_head(const float* __restrict__ f1, const float* __restrict__ f2,
    const float* __restrict__ fw1, const float* __restrict__ fb1,
    const float* __restrict__ fw2, const float* __restrict__ fb2,
    const float* __restrict__ fc1w, const float* __restrict__ fc1b,
    const float* __restrict__ fc2w, const float* __restrict__ fc2b,
    const float* __restrict__ prw, const float* __restrict__ prb,
    float* __restrict__ out){
  __shared__ float cat[2*H_], a1[128], fused[H_], b1[512], b2[512], red[256];
  __shared__ float wv[2];
  int b = blockIdx.x, t = threadIdx.x;
  cat[t] = f1[b*H_+t]; cat[H_+t] = f2[b*H_+t];
  __syncthreads();
  if (t<128){ float s=fb1[t]; for(int j=0;j<2*H_;j++) s+=cat[j]*fw1[t*2*H_+j]; a1[t]=fmaxf(s,0.f); }
  __syncthreads();
  if (t<2){ float s=fb2[t]; for(int j=0;j<128;j++) s+=a1[j]*fw2[t*128+j]; wv[t]=sigmoidf_(s); }
  __syncthreads();
  fused[t] = wv[0]*cat[t] + wv[1]*cat[H_+t];
  __syncthreads();
  for (int o=t;o<512;o+=256){ float s=fc1b[o]; for(int j=0;j<H_;j++) s+=fused[j]*fc1w[o*H_+j]; b1[o]=fmaxf(s,0.f); }
  __syncthreads();
  for (int o=t;o<512;o+=256){ float s=fc2b[o]; for(int j=0;j<512;j++) s+=b1[j]*fc2w[o*512+j]; b2[o]=fmaxf(s,0.f); }
  __syncthreads();
  red[t] = b2[t]*prw[t] + b2[t+256]*prw[t+256];
  __syncthreads();
  for (int off=128; off>0; off>>=1){ if (t<off) red[t]+=red[t+off]; __syncthreads(); }
  if (t==0) out[b] = red[0] + prb[0];
}

// ---------------- host orchestration ----------------
// Workspace needed: ~275 MB.

extern "C" void kernel_launch(void* const* d_in, const int* in_sizes, int n_in,
                              void* d_out, int out_size, void* d_ws, size_t ws_size,
                              hipStream_t stream){
  (void)in_sizes; (void)n_in; (void)out_size; (void)ws_size;
  const float* x = (const float*)d_in[0];
  const float* A = (const float*)d_in[1];
  const float* fw1  = (const float*)d_in[38];
  const float* fb1  = (const float*)d_in[39];
  const float* fw2  = (const float*)d_in[40];
  const float* fb2  = (const float*)d_in[41];
  const float* fc1w = (const float*)d_in[42];
  const float* fc1b = (const float*)d_in[43];
  const float* fc2w = (const float*)d_in[44];
  const float* fc2b = (const float*)d_in[45];
  const float* prw  = (const float*)d_in[46];
  const float* prb  = (const float*)d_in[47];

  char* p = (char*)d_ws;
  auto alloc = [&](size_t bytes)->char*{
    char* r = p; p += (bytes + 255) & ~(size_t)255; return r;
  };
  float* deg    = (float*)alloc((size_t)ROWS*4);
  float* rowsum = (float*)alloc((size_t)ROWS*4);
  float* f1     = (float*)alloc((size_t)B_*H_*4);
  float* f2     = (float*)alloc((size_t)B_*H_*4);
  float* h1     = (float*)alloc((size_t)ROWS*H_*4);
  float* h2     = (float*)alloc((size_t)ROWS*H_*4);
  _Float16* xg   = (_Float16*)alloc((size_t)ROWS*H_*2);
  _Float16* xgT  = (_Float16*)alloc((size_t)ROWS*H_*2);
  _Float16* qkb  = (_Float16*)alloc((size_t)ROWS*H_*2);
  _Float16* Z1   = (_Float16*)alloc((size_t)ROWS*H_*2);
  _Float16* zln  = (_Float16*)alloc((size_t)ROWS*H_*2);
  _Float16* S    = (_Float16*)alloc((size_t)B_*N_*N_*2);
  _Float16* zup  = (_Float16*)alloc((size_t)ROWS*FH_*2);
  _Float16* qwT16  = (_Float16*)alloc((size_t)H_*H_*2);
  _Float16* lw16   = (_Float16*)alloc((size_t)H_*H_*2);
  _Float16* rw16   = (_Float16*)alloc((size_t)H_*H_*2);
  _Float16* upwT16 = (_Float16*)alloc((size_t)G_*128*32*2);
  _Float16* dnwT16 = (_Float16*)alloc((size_t)G_*32*128*2);

  k_deg<<<ROWS/8, 256, 0, stream>>>(A, deg);

  for (int l=0;l<2;l++){
    int d = (l==0) ? F0_ : H_;
    const float* xin = (l==0) ? x : h1;
    float* hcur = (l==0) ? h1 : h2;
    int base = 2 + l*18;
    const float* qk_w=(const float*)d_in[base+0];  const float* qk_b=(const float*)d_in[base+1];
    const float* d_w =(const float*)d_in[base+2];  const float* d_b =(const float*)d_in[base+3];
    const float* ll_w=(const float*)d_in[base+4];  const float* ll_b=(const float*)d_in[base+5];
    const float* lr_w=(const float*)d_in[base+6];
    const float* ln_g=(const float*)d_in[base+7];  const float* ln_b=(const float*)d_in[base+8];
    const float* up_w=(const float*)d_in[base+9];  const float* up_b=(const float*)d_in[base+10];
    const float* up_g=(const float*)d_in[base+11]; const float* up_be=(const float*)d_in[base+12];
    const float* dn_w=(const float*)d_in[base+13]; const float* dn_b=(const float*)d_in[base+14];
    const float* dn_g=(const float*)d_in[base+15]; const float* dn_be=(const float*)d_in[base+16];
    const float* ls  =(const float*)d_in[base+17];

    k_f32_to_f16<<<(d*d+255)/256,256,0,stream>>>(qk_w, qwT16, d*d);
    k_f32_to_f16<<<(H_*d+255)/256,256,0,stream>>>(ll_w, lw16, H_*d);
    k_f32_to_f16<<<(H_*d+255)/256,256,0,stream>>>(lr_w, rw16, H_*d);
    k_glt_wT<<<(G_*32*128+255)/256,256,0,stream>>>(up_w, upwT16, 32, 128);
    k_glt_wT<<<(G_*128*32+255)/256,256,0,stream>>>(dn_w, dnwT16, 128, 32);

    k_gate<<<(ROWS*d)/256, 256, 0, stream>>>(xin, deg, d_w, d_b, xg, xgT, d);
    k_qk<<<dim3(ROWS/16, d/128), 256, 0, stream>>>(xg, qwT16, qk_b, qkb, d);
    k_zero<<<ROWS/256, 256, 0, stream>>>(rowsum, ROWS);
    float scale = 1.0f / sqrtf((float)d);
    k_score<<<dim3(B_, 8, 8), 256, 0, stream>>>(qkb, A, S, rowsum, scale, d);
    k_attnx<<<dim3(B_, 8, d/128), 256, 0, stream>>>(S, xgT, rowsum, Z1, d);
    k_mix<<<ROWS/16, 256, 0, stream>>>(Z1, xg, lw16, rw16, ll_b, ln_g, ln_b, hcur, zln, d);
    k_glt_up<<<ROWS/16, 256, 0, stream>>>(zln, upwT16, up_b, up_g, up_be, zup);
    k_glt_dn<<<ROWS/16, 256, 0, stream>>>(zup, dnwT16, dn_b, dn_g, dn_be, ls, hcur);
  }

  k_pool<<<B_, H_, 0, stream>>>(h1, f1);
  k_pool<<<B_, H_, 0, stream>>>(h2, f2);
  k_head<<<B_, 256, 0, stream>>>(f1, f2, fw1, fb1, fw2, fb2,
                                 fc1w, fc1b, fc2w, fc2b, prw, prb, (float*)d_out);
}
